// HTA_9526237462866
// MI455X (gfx1250) — matrix operand order repaired and checked by reading, not compiled, not run on hardware
//
#include <hip/hip_runtime.h>
#include <math.h>

// ---------------------------------------------------------------------------
// Problem constants (from reference)
// ---------------------------------------------------------------------------
#define BN    8
#define CC    192
#define HH    128
#define WW    128
#define NHEAD 4
#define CPER  (CC / NHEAD)          // 48
#define DDIM  (CPER * HH)           // 6144
#define HW    (HH * WW)             // 16384

typedef __attribute__((ext_vector_type(16))) __bf16 v16bf;
typedef __attribute__((ext_vector_type(8)))  float  v8f;

#ifndef __has_builtin
#define __has_builtin(x) 0
#endif
#if __has_builtin(__builtin_amdgcn_tensor_load_to_lds) && \
    __has_builtin(__builtin_amdgcn_s_wait_tensorcnt)
#define USE_TDM 1
#else
#define USE_TDM 0
#endif

#if USE_TDM
typedef __attribute__((ext_vector_type(4))) unsigned int u32x4;
typedef __attribute__((ext_vector_type(8))) int          i32x8;
typedef __attribute__((ext_vector_type(4))) int          i32x4;

// Issue a TDM 2D tile load: 16 rows x 128 f32 (contiguous rows, row stride =
// WW elements) from global -> LDS.  D# per CDNA5 ISA ch.8:
//  group0: [1:0]=count=1, [63:32]=lds_addr, [120:64]=global_addr, [127:126]=2
//  group1: data_size=4B, tensor_dim0=128, tensor_dim1=DDIM, tile=128x16,
//          tensor_dim0_stride=128
// This toolchain exposes the 6-arg builtin:
//  (u32x4 g0, i32x8 g1, i32x4 g2, i32x4 g3, i32x8 extra, i32 cpol)
__device__ __forceinline__ void tdm_load_vtile(unsigned lds_off, const float* gptr) {
    unsigned long long ga = (unsigned long long)(size_t)gptr;
    u32x4 g0;
    g0[0] = 1u;                                               // count=1, user mode
    g0[1] = lds_off;                                          // lds_addr
    g0[2] = (unsigned)ga;                                     // global_addr[31:0]
    g0[3] = ((unsigned)(ga >> 32) & 0x01FFFFFFu) | (2u << 30); // [56:32] | type=2
    i32x8 g1;
    g1[0] = (int)(2u << 16);          // workgroup_mask=0, data_size=2 (4 bytes)
    g1[1] = (int)(128u << 16);        // tensor_dim0[15:0] = 128
    g1[2] = (int)((unsigned)DDIM << 16); // tensor_dim1[15:0] = 6144
    g1[3] = (int)(128u << 16);        // tile_dim0 = 128
    g1[4] = 16;                       // tile_dim1 = 16 (tile_dim2 = 0)
    g1[5] = 128;                      // tensor_dim0_stride = 128
    g1[6] = 0;                        // tensor_dim1_stride = 0 (2D)
    g1[7] = 0;
    i32x4 z4 = {0, 0, 0, 0};
    i32x8 z8 = {0, 0, 0, 0, 0, 0, 0, 0};
    __builtin_amdgcn_tensor_load_to_lds(g0, g1, z4, z4, z8, 0);
}
#endif

// ---------------------------------------------------------------------------
// Kernel 1/5: pointwise (1x1) conv as bf16 WMMA GEMM with f32 accumulate.
// out[b, o, p] = sum_c in[b, c, p] * w[o*Cin + c]
// One wave computes a 16(M=Cout) x 16(N=pixels) tile, K-loop over Cin in 32s.
// ---------------------------------------------------------------------------
__global__ __launch_bounds__(32)
void pw_conv_wmma(const float* __restrict__ in, const float* __restrict__ w,
                  float* __restrict__ out, int Cin) {
    const int lane  = threadIdx.x & 31;
    const int n0    = blockIdx.x * 16;      // flat pixel tile over B*HW (HW%16==0)
    const int m0    = blockIdx.y * 16;      // Cout tile
    const int b     = n0 / HW;
    const int p0    = n0 % HW;
    const int mrow  = lane & 15;
    const int ksel  = (lane >> 4) << 3;     // 0 or 8 (A layout)
    const int khalf = (lane >> 4) << 4;     // 0 or 16 (unused; B also uses ksel)
    const float* inb = in + (size_t)b * Cin * HW;

    v8f acc = {};
    const int col = p0 + mrow;
    for (int k0 = 0; k0 < Cin; k0 += 32) {
        v16bf A, Bm;
#pragma unroll
        for (int j = 0; j < 8; ++j) {
            int K = ((j >> 2) << 4) + ksel + ((j & 3) << 1);
            A[2 * j]     = (__bf16)w[(m0 + mrow) * Cin + k0 + K];
            A[2 * j + 1] = (__bf16)w[(m0 + mrow) * Cin + k0 + K + 1];
        }
#pragma unroll
        for (int j = 0; j < 8; ++j) {
            int kk = k0 + ((j >> 2) << 4) + ksel + ((j & 3) << 1);
            Bm[2 * j]     = (__bf16)inb[(size_t)kk * HW + col];
            Bm[2 * j + 1] = (__bf16)inb[(size_t)(kk + 1) * HW + col];
        }
        acc = __builtin_amdgcn_wmma_f32_16x16x32_bf16(false, A, false, Bm,
                                                      (short)0, acc, false, false);
    }
    const int rowoff = (lane >> 4) << 3;
    float* outb = out + (size_t)b * gridDim.y * 16 * HW;
#pragma unroll
    for (int r = 0; r < 8; ++r)
        outb[(size_t)(m0 + r + rowoff) * HW + col] = acc[r];
}

// ---------------------------------------------------------------------------
// Kernel 2/5: depthwise 3x3 conv, pad=1, groups = 3C. Memory bound; scalar f32.
// ---------------------------------------------------------------------------
__global__ __launch_bounds__(256)
void dw3x3(const float* __restrict__ in, const float* __restrict__ wdw,
           float* __restrict__ out, long total) {
    long idx = (long)blockIdx.x * blockDim.x + threadIdx.x;
    if (idx >= total) return;
    const int wq = (int)(idx & (WW - 1));
    const int hq = (int)((idx >> 7) & (HH - 1));
    const long chb = idx >> 14;             // b*3C + ch
    const int ch = (int)(chb % (3 * CC));
    const float* ip = in + (chb << 14);
    const float* wp = wdw + ch * 9;
    float s = 0.f;
#pragma unroll
    for (int dy = -1; dy <= 1; ++dy) {
        int hh = hq + dy;
        if (hh < 0 || hh >= HH) continue;
#pragma unroll
        for (int dx = -1; dx <= 1; ++dx) {
            int ww2 = wq + dx;
            if (ww2 < 0 || ww2 >= WW) continue;
            s += ip[hh * WW + ww2] * wp[(dy + 1) * 3 + dx + 1];
        }
    }
    out[idx] = s;
}

// ---------------------------------------------------------------------------
// Kernel 3/5: row L2 norms of q and k.  inv_norm[which][b][head][w].
// q/k element (d, w) sits at base[d*W + w] since d = c*H + h matches strides.
// ---------------------------------------------------------------------------
__global__ __launch_bounds__(256)
void rownorm(const float* __restrict__ qkv, float* __restrict__ inv_norm) {
    const int wq = blockIdx.x, n = blockIdx.y;
    const int b = blockIdx.z >> 1, which = blockIdx.z & 1;
    const float* base = qkv + (size_t)(b * 3 * CC + which * CC + n * CPER) * HW;
    float s = 0.f;
    for (int d = threadIdx.x; d < DDIM; d += 256) {
        float v = base[(size_t)d * WW + wq];
        s += v * v;
    }
#pragma unroll
    for (int off = 16; off >= 1; off >>= 1) s += __shfl_xor(s, off, 32);
    __shared__ float red[8];
    if ((threadIdx.x & 31) == 0) red[threadIdx.x >> 5] = s;
    __syncthreads();
    if (threadIdx.x == 0) {
        float t = 0.f;
#pragma unroll
        for (int i = 0; i < 8; ++i) t += red[i];
        inv_norm[(((size_t)which * BN + b) * NHEAD + n) * WW + wq] =
            1.0f / fmaxf(sqrtf(t), 1e-12f);
    }
}

// ---------------------------------------------------------------------------
// Kernel 4/5: attention per (b, head). Block = 256 threads = 8 waves.
// Phase 1: S = q_hat k_hat^T, k chunk staged pair-packed bf16 in LDS.
// Softmax via half-wave shfl_xor reductions.
// Phase 2: out = P v, v tile (16 d x 128 w', f32) fetched by the Tensor Data
//          Mover into a double-buffered LDS tile, overlapping WMMA compute;
//          fenced with s_wait_tensorcnt + barrier.
// LDS: kbufp 8KB | pbuf 64KB | vbuf0 8KB | vbuf1 8KB  = 88KB.
// ---------------------------------------------------------------------------
__global__ __launch_bounds__(256)
void attn_kernel(const float* __restrict__ qkv, const float* __restrict__ inv_norm,
                 const float* __restrict__ temperature, float* __restrict__ attn_out) {
    extern __shared__ char smem[];
    unsigned int* kbufp = (unsigned int*)smem;                 // 16*128 u32 = 8KB
    float*        pbuf  = (float*)(smem + 8192);               // 128*128 f32 = 64KB
    float*        vbuf0 = (float*)(smem + 8192 + 65536);       // 16*128 f32 = 8KB
    float*        vbuf1 = (float*)(smem + 8192 + 65536 + 8192);// 16*128 f32 = 8KB

    const int b = blockIdx.x / NHEAD;
    const int n = blockIdx.x % NHEAD;
    const float* qg = qkv + (size_t)(b * 3 * CC + 0 * CC + n * CPER) * HW;
    const float* kg = qkv + (size_t)(b * 3 * CC + 1 * CC + n * CPER) * HW;
    const float* vg = qkv + (size_t)(b * 3 * CC + 2 * CC + n * CPER) * HW;
    const float* invq = inv_norm + ((size_t)(0 * BN + b) * NHEAD + n) * WW;
    const float* invk = inv_norm + ((size_t)(1 * BN + b) * NHEAD + n) * WW;
    float* ab = attn_out + (size_t)(b * CC + n * CPER) * HW;

    const int lane   = threadIdx.x & 31;
    const int wave   = threadIdx.x >> 5;
    const int mrow   = lane & 15;
    const int ksel   = (lane >> 4) << 3;   // A layout K select
    const int kpbase = (lane >> 4) << 3;   // B layout pair base (0 or 8)
    const int rowoff = (lane >> 4) << 3;   // C/D layout row offset
    const int wrow   = wave * 16 + mrow;   // this lane's query row (A/M)
    const float invq_w = invq[wrow];

    // ---------------- Phase 1: scores ----------------
    v8f acc[8] = {};
    for (int d0 = 0; d0 < DDIM; d0 += 32) {
        // Stage normalized k chunk, pair-packed: kbufp[kp*128+col] holds
        // bf16(K=2kp) | bf16(K=2kp+1)<<16  (matches B-operand reg layout).
        for (int i = threadIdx.x; i < 16 * 128; i += 256) {
            int kp = i >> 7, colc = i & 127;
            float ik = invk[colc];
            union { __bf16 h[2]; unsigned int u; } P;
            P.h[0] = (__bf16)(kg[(size_t)(d0 + 2 * kp) * WW + colc] * ik);
            P.h[1] = (__bf16)(kg[(size_t)(d0 + 2 * kp + 1) * WW + colc] * ik);
            kbufp[i] = P.u;
        }
        __syncthreads();
        v16bf A;
#pragma unroll
        for (int j = 0; j < 8; ++j) {
            int K = ((j >> 2) << 4) + ksel + ((j & 3) << 1);
            A[2 * j]     = (__bf16)(qg[(size_t)(d0 + K) * WW + wrow] * invq_w);
            A[2 * j + 1] = (__bf16)(qg[(size_t)(d0 + K + 1) * WW + wrow] * invq_w);
        }
#pragma unroll
        for (int t = 0; t < 8; ++t) {
            union { v16bf v; unsigned int u[8]; } Bu;
            int colc = (t << 4) + mrow;
#pragma unroll
            for (int j = 0; j < 8; ++j)
                Bu.u[j] = kbufp[(((j >> 2) << 3) + (kpbase >> 1) + (j & 3)) * 128 + colc];
            acc[t] = __builtin_amdgcn_wmma_f32_16x16x32_bf16(false, A, false, Bu.v,
                                                             (short)0, acc[t], false, false);
        }
        __syncthreads();
    }

    // ---------------- Softmax (row = wave*16 + r + rowoff) ----------------
    const float temp = temperature[n];
#pragma unroll
    for (int r = 0; r < 8; ++r) {
        float m = -3.4e38f;
#pragma unroll
        for (int t = 0; t < 8; ++t) { acc[t][r] *= temp; m = fmaxf(m, acc[t][r]); }
#pragma unroll
        for (int off = 8; off >= 1; off >>= 1) m = fmaxf(m, __shfl_xor(m, off, 32));
        float s = 0.f;
#pragma unroll
        for (int t = 0; t < 8; ++t) {
            float e = __expf(acc[t][r] - m);
            acc[t][r] = e;
            s += e;
        }
#pragma unroll
        for (int off = 8; off >= 1; off >>= 1) s += __shfl_xor(s, off, 32);
        float inv = 1.0f / s;
#pragma unroll
        for (int t = 0; t < 8; ++t)
            pbuf[(wave * 16 + r + rowoff) * 128 + (t << 4) + mrow] = acc[t][r] * inv;
    }
    __syncthreads();

    // ---------------- Phase 2: out = P @ v ----------------
#if USE_TDM
    const unsigned voff0 = (unsigned)(size_t)(void*)vbuf0;
    const unsigned voff1 = (unsigned)(size_t)(void*)vbuf1;
    if (wave == 0) {                       // whole-wave issue, EXEC all ones
        tdm_load_vtile(voff0, vg);
        __builtin_amdgcn_s_wait_tensorcnt(0);
    }
    __syncthreads();
#endif
    for (int dt = 0; dt < DDIM / 16; ++dt) {
        const float* vcur;
#if USE_TDM
        vcur = (dt & 1) ? vbuf1 : vbuf0;
        if (wave == 0 && dt + 1 < DDIM / 16)   // prefetch next tile via TDM
            tdm_load_vtile((dt & 1) ? voff0 : voff1,
                           vg + (size_t)(dt + 1) * 16 * WW);
#else
        vcur = vbuf0;
        for (int i = threadIdx.x; i < 16 * 128; i += 256)
            vbuf0[i] = vg[(size_t)(dt * 16 + (i >> 7)) * WW + (i & 127)];
        __syncthreads();
#endif
        v8f o = {};
        const float* prow = pbuf + wrow * 128;
        const float* vrow = vcur + mrow * 128;   // B col (d) = mrow
#pragma unroll
        for (int ch = 0; ch < 4; ++ch) {         // K = 128 over w'
            v16bf A, Bm;
#pragma unroll
            for (int j = 0; j < 8; ++j) {
                int K = (ch << 5) + ((j >> 2) << 4) + ksel + ((j & 3) << 1);
                A[2 * j]     = (__bf16)prow[K];
                A[2 * j + 1] = (__bf16)prow[K + 1];
            }
#pragma unroll
            for (int j = 0; j < 8; ++j) {
                int kp = (ch << 4) + ((j >> 2) << 3) + (kpbase >> 1) + (j & 3); // w' pair index
                const float2 f = *(const float2*)(vrow + 2 * kp); // ds_load_b64
                Bm[2 * j]     = (__bf16)f.x;
                Bm[2 * j + 1] = (__bf16)f.y;
            }
            o = __builtin_amdgcn_wmma_f32_16x16x32_bf16(false, A, false, Bm,
                                                        (short)0, o, false, false);
        }
        const int d = dt * 16 + mrow;            // D layout: N -> d col
#pragma unroll
        for (int r = 0; r < 8; ++r)
            ab[(size_t)d * WW + (wave * 16 + r + rowoff)] = o[r];
#if USE_TDM
        if (wave == 0) __builtin_amdgcn_s_wait_tensorcnt(0);
#endif
        __syncthreads();
    }
}

// ---------------------------------------------------------------------------
// Host launcher
// Inputs: 0:x  1:w_qkv  2:w_dw  3:w_proj  4:temperature
// Workspace: qkv_pre | qkv | inv_norm ; attn_out aliases qkv_pre.
// ---------------------------------------------------------------------------
extern "C" void kernel_launch(void* const* d_in, const int* in_sizes, int n_in,
                              void* d_out, int out_size, void* d_ws, size_t ws_size,
                              hipStream_t stream) {
    const float* x           = (const float*)d_in[0];
    const float* w_qkv       = (const float*)d_in[1];
    const float* w_dw        = (const float*)d_in[2];
    const float* w_proj      = (const float*)d_in[3];
    const float* temperature = (const float*)d_in[4];
    float* out = (float*)d_out;

    const size_t qkv_elems = (size_t)BN * 3 * CC * HW;
    float* qkv_pre  = (float*)d_ws;
    float* qkv      = qkv_pre + qkv_elems;
    float* inv_norm = qkv + qkv_elems;
    float* attn_out = qkv_pre;   // alias: qkv_pre dead after dw conv

    {   // 1) qkv 1x1 conv (GEMM 576 x 131072 x 192)
        dim3 grid((BN * HW) / 16, (3 * CC) / 16);
        pw_conv_wmma<<<grid, 32, 0, stream>>>(x, w_qkv, qkv_pre, CC);
    }
    {   // 2) depthwise 3x3
        long total = (long)BN * 3 * CC * HW;
        dw3x3<<<(int)((total + 255) / 256), 256, 0, stream>>>(qkv_pre, w_dw, qkv, total);
    }
    {   // 3) q/k row inverse norms
        dim3 grid(WW, NHEAD, BN * 2);
        rownorm<<<grid, 256, 0, stream>>>(qkv, inv_norm);
    }
    {   // 4) attention
        size_t shmem = 8192 + 65536 + 8192 + 8192;   // 88KB
        attn_kernel<<<BN * NHEAD, 256, shmem, stream>>>(qkv, inv_norm, temperature, attn_out);
    }
    {   // 5) proj 1x1 conv (GEMM 192 x 131072 x 192)
        dim3 grid((BN * HW) / 16, CC / 16);
        pw_conv_wmma<<<grid, 32, 0, stream>>>(attn_out, w_proj, out, CC);
    }
}
